// CPWanSelfAttention_78838419685970
// MI455X (gfx1250) — matrix operand order repaired and checked by reading, not compiled, not run on hardware
//
#include <hip/hip_runtime.h>
#include <hip/hip_bf16.h>

typedef _Float16 v4h  __attribute__((ext_vector_type(4)));
typedef _Float16 v8h  __attribute__((ext_vector_type(8)));
typedef _Float16 v16h __attribute__((ext_vector_type(16)));
typedef float    v8f  __attribute__((ext_vector_type(8)));

#define DIM   2048
#define SEQ   1992
#define SPAD  2048
#define HEADS 16
#define DHEAD 128
#define LP    40   // LDS tile pitch in halves (32 data + 8 pad vs bank conflicts)

// ---------------------------------------------------------------------------
// WMMA helper (CDNA5 v_wmma_f32_16x16x32_f16)
// ---------------------------------------------------------------------------
__device__ __forceinline__ v8f wmma_f16(v16h a, v16h b, v8f c) {
  return __builtin_amdgcn_wmma_f32_16x16x32_f16(false, a, false, b, (short)0, c,
                                                false, false);
}

// A-matrix 16x32 f16 fragment from row-major [row][k] (pitch ld halves).
// lanes 0-15 : row=lane,    K = {0..7, 16..23}; lanes 16-31: K = {8..15, 24..31}
__device__ __forceinline__ v16h load_frag_a(const _Float16* base, int ld, int lane) {
  const _Float16* p = base + (lane & 15) * ld + ((lane >> 4) << 3);
  v8h lo = *(const v8h*)(p);
  v8h hi = *(const v8h*)(p + 16);
  return __builtin_shufflevector(lo, hi, 0, 1, 2, 3, 4, 5, 6, 7,
                                 8, 9, 10, 11, 12, 13, 14, 15);
}

// B-matrix 32x16 f16 fragment from row-major [n][k]: lane holds col n=lane&15,
// lanes 0-15: K = {0..7, 16..23}; lanes 16-31: K = {8..15, 24..31}. Two 16B loads.
__device__ __forceinline__ v16h load_frag_b(const _Float16* base, int ld, int lane) {
  const _Float16* p = base + (lane & 15) * ld + ((lane >> 4) << 3);
  v8h lo = *(const v8h*)(p);
  v8h hi = *(const v8h*)(p + 16);
  return __builtin_shufflevector(lo, hi, 0, 1, 2, 3, 4, 5, 6, 7,
                                 8, 9, 10, 11, 12, 13, 14, 15);
}

// ---------------------------------------------------------------------------
// CDNA5 async global -> LDS copy (16B per lane), tracked with ASYNCcnt.
// VDST = AS(3) pointer (32-bit LDS byte address), VADDR = 64-bit global addr.
// ---------------------------------------------------------------------------
__device__ __forceinline__ void async_copy_b128(_Float16* lds_generic,
                                                const _Float16* gptr) {
  __attribute__((address_space(3))) _Float16* l =
      (__attribute__((address_space(3))) _Float16*)lds_generic;
  asm volatile("global_load_async_to_lds_b128 %0, %1, off"
               :
               : "v"(l), "v"(gptr)
               : "memory");
}

template <int N>
__device__ __forceinline__ void wait_asynccnt() {
  asm volatile("s_wait_asynccnt %0" : : "i"(N) : "memory");
}

// ---------------------------------------------------------------------------
// Kernel 0: f32 -> f16 packing (X padded to 2048 rows, 4 weight matrices)
// ---------------------------------------------------------------------------
__global__ void __launch_bounds__(256) pack_f16_kernel(
    const float* __restrict__ hs, const float* __restrict__ wq,
    const float* __restrict__ wk, const float* __restrict__ wv,
    const float* __restrict__ wo,
    _Float16* __restrict__ x16, _Float16* __restrict__ wq16,
    _Float16* __restrict__ wk16, _Float16* __restrict__ wv16,
    _Float16* __restrict__ wo16) {
  const int z = blockIdx.y;
  const long idx = ((long)blockIdx.x * 256 + threadIdx.x) * 4;  // over 2048*2048
  const float* src;
  _Float16* dst;
  if (z == 0)      { src = hs; dst = x16; }
  else if (z == 1) { src = wq; dst = wq16; }
  else if (z == 2) { src = wk; dst = wk16; }
  else if (z == 3) { src = wv; dst = wv16; }
  else             { src = wo; dst = wo16; }
  v4h out = {};
  if (!(z == 0 && idx >= (long)SEQ * DIM)) {
    const float4 v = *(const float4*)(src + idx);
    out[0] = (_Float16)v.x; out[1] = (_Float16)v.y;
    out[2] = (_Float16)v.z; out[3] = (_Float16)v.w;
  }
  *(v4h*)(dst + idx) = out;
}

// ---------------------------------------------------------------------------
// Kernel 1: GEMM  out = X (MxK) · Wᵀ (NxK row-major) + bias
// Block = 4 waves, 128x128 tile; A/B k-slabs (128x32 f16) staged to LDS by
// double-buffered GLOBAL_LOAD_ASYNC_TO_LDS_B128 (ASYNCcnt). Each wave does a
// 64x64 quadrant as 4x4 WMMA tiles.
// VARIANT 0: f16 out[m][n], zero for m>=Mvalid      (Q, K)
// VARIANT 1: f16 out[n][m] (transposed), zero pad   (Vᵀ)
// VARIANT 2: f32 out[m][n], store only m<Mvalid     (final projection)
// ---------------------------------------------------------------------------
template <int VARIANT>
__global__ void __launch_bounds__(128) gemm_xwT_kernel(
    const _Float16* __restrict__ X, const _Float16* __restrict__ W,
    const float* __restrict__ bias, _Float16* __restrict__ out16,
    float* __restrict__ outf, int Mvalid) {
  __shared__ __attribute__((aligned(16))) _Float16 ldsA[2][128 * LP];
  __shared__ __attribute__((aligned(16))) _Float16 ldsB[2][128 * LP];

  const int t = threadIdx.x;       // 0..127
  const int lane = t & 31;
  const int wv = t >> 5;           // wave id 0..3
  const int mq = (wv & 1) * 64;    // wave quadrant inside 128x128 tile
  const int nq = (wv >> 1) * 64;
  const int m_blk = blockIdx.x * 128;
  const int n_blk = blockIdx.y * 128;

  // Stage one 128x32 k-slab of A (from X) and B (from W) into LDS buffer `buf`.
  // 1024 16B chunks total, 8 async copies per thread (4 A + 4 B).
  auto stage = [&](int buf, int k0) {
#pragma unroll
    for (int i = 0; i < 4; ++i) {
      const int c = i * 128 + t;     // 0..511
      const int row = c >> 2;        // 0..127
      const int q = c & 3;           // 16B quarter of the 64B row
      async_copy_b128(&ldsA[buf][row * LP + q * 8],
                      X + (long)(m_blk + row) * DIM + k0 + q * 8);
      async_copy_b128(&ldsB[buf][row * LP + q * 8],
                      W + (long)(n_blk + row) * DIM + k0 + q * 8);
    }
  };

  v8f acc[4][4] = {};
  stage(0, 0);
  for (int kt = 0; kt < DIM / 32; ++kt) {
    const int cur = kt & 1;
    if (kt + 1 < DIM / 32) {
      stage(cur ^ 1, (kt + 1) * 32);
      wait_asynccnt<8>();   // own 8 copies of `cur` complete (in-order)
    } else {
      wait_asynccnt<0>();
    }
    __syncthreads();        // all waves' `cur` slab visible in LDS

    v16h a[4], b[4];
#pragma unroll
    for (int i = 0; i < 4; ++i)
      a[i] = load_frag_a(&ldsA[cur][(mq + 16 * i) * LP], LP, lane);
#pragma unroll
    for (int j = 0; j < 4; ++j)
      b[j] = load_frag_b(&ldsB[cur][(nq + 16 * j) * LP], LP, lane);
#pragma unroll
    for (int i = 0; i < 4; ++i)
#pragma unroll
      for (int j = 0; j < 4; ++j)
        acc[i][j] = wmma_f16(a[i], b[j], acc[i][j]);

    __syncthreads();        // reads done before `cur` is overwritten
  }

  const int rbase = (lane >> 4) * 8;  // D layout: VGPR r -> M = r + 8*half
  const int nc = lane & 15;
#pragma unroll
  for (int i = 0; i < 4; ++i) {
#pragma unroll
    for (int r = 0; r < 8; ++r) {
      const int m = m_blk + mq + 16 * i + rbase + r;
      const bool valid = (m < Mvalid);
#pragma unroll
      for (int j = 0; j < 4; ++j) {
        const int n = n_blk + nq + 16 * j + nc;
        const float v = acc[i][j][r] + bias[n];
        if (VARIANT == 0) {
          out16[(long)m * DIM + n] = valid ? (_Float16)v : (_Float16)0.f;
        } else if (VARIANT == 1) {
          out16[(long)n * SPAD + m] = valid ? (_Float16)v : (_Float16)0.f;
        } else {
          if (valid) outf[(long)m * DIM + n] = v;
        }
      }
    }
  }
}

// ---------------------------------------------------------------------------
// Kernel 2: RMSNorm(full DIM) + RoPE (interleaved pairs), in place on f16 Q/K
// ---------------------------------------------------------------------------
__global__ void __launch_bounds__(256) rmsnorm_rope_kernel(
    _Float16* __restrict__ q16, _Float16* __restrict__ k16,
    const float* __restrict__ nw_q, const float* __restrict__ nw_k,
    const float* __restrict__ fcos, const float* __restrict__ fsin) {
  const int s = blockIdx.x;
  _Float16* mat = blockIdx.y ? k16 : q16;
  const float* w = blockIdx.y ? nw_k : nw_q;
  const int t = threadIdx.x;
  __shared__ float red[8];

  float x[8];
  float ss = 0.f;
#pragma unroll
  for (int e = 0; e < 8; ++e) {
    x[e] = (float)mat[(long)s * DIM + t * 8 + e];
    ss += x[e] * x[e];
  }
#pragma unroll
  for (int o = 16; o > 0; o >>= 1) ss += __shfl_xor(ss, o, 32);
  if ((t & 31) == 0) red[t >> 5] = ss;
  __syncthreads();
  float tot = 0.f;
#pragma unroll
  for (int i = 0; i < 8; ++i) tot += red[i];
  const float inv = rsqrtf(tot / (float)DIM + 1e-6f);

#pragma unroll
  for (int e = 0; e < 8; e += 2) {
    const int g = t * 8 + e;        // even global dim
    const int p = g & (DHEAD - 1);  // even index within head
    const float c  = fcos[(long)s * DHEAD + p];
    const float sn = fsin[(long)s * DHEAD + p + 1];
    const float x1 = x[e]     * inv * w[g];
    const float x2 = x[e + 1] * inv * w[g + 1];
    mat[(long)s * DIM + g]     = (_Float16)(x1 * c - x2 * sn);
    mat[(long)s * DIM + g + 1] = (_Float16)(x1 * sn + x2 * c);
  }
}

// ---------------------------------------------------------------------------
// Kernel 3: attention, one wave per (16-query tile, head).
// Scores strip 16x2048 f32 in LDS -> softmax -> P f16 in LDS -> PV WMMA.
// ---------------------------------------------------------------------------
__global__ void __launch_bounds__(32) attention_kernel(
    const _Float16* __restrict__ Q16, const _Float16* __restrict__ K16,
    const _Float16* __restrict__ Vt, _Float16* __restrict__ O16) {
  extern __shared__ float smem[];
  float* sc = smem;                             // 16 x 2048 f32
  _Float16* P = (_Float16*)(smem + 16 * SPAD);  // 16 x 2048 f16

  const int lane = threadIdx.x;
  const int q0 = blockIdx.x * 16;
  const int h  = blockIdx.y;
  const int half = lane >> 4;
  const int nc = lane & 15;

  v16h aq[4];
#pragma unroll
  for (int kk = 0; kk < 4; ++kk)
    aq[kk] = load_frag_a(Q16 + (long)q0 * DIM + h * DHEAD + kk * 32, DIM, lane);

  // ---- S = Q·Kᵀ ----
  for (int nt = 0; nt < SPAD / 16; ++nt) {
    v8f s = {};
#pragma unroll
    for (int kk = 0; kk < 4; ++kk) {
      v16h bk = load_frag_b(K16 + (long)(nt * 16) * DIM + h * DHEAD + kk * 32,
                            DIM, lane);
      s = wmma_f16(aq[kk], bk, s);
    }
#pragma unroll
    for (int r = 0; r < 8; ++r)
      sc[(half * 8 + r) * SPAD + nt * 16 + nc] = s[r];
  }
  __syncthreads();

  // ---- softmax over 2048 keys (scale folded into exp) ----
  const float scale = 0.08838834764831845f;  // 1/sqrt(128)
  const int row = lane & 15;
  const int c0 = half * (SPAD / 2);
  float mx = -3.4e38f;
  for (int c = 0; c < SPAD / 2; ++c) mx = fmaxf(mx, sc[row * SPAD + c0 + c]);
  mx = fmaxf(mx, __shfl_xor(mx, 16, 32));
  float sum = 0.f;
  for (int c = 0; c < SPAD / 2; ++c) {
    const float e = __expf((sc[row * SPAD + c0 + c] - mx) * scale);
    sc[row * SPAD + c0 + c] = e;
    sum += e;
  }
  sum += __shfl_xor(sum, 16, 32);
  const float invs = 1.f / sum;
  for (int c = 0; c < SPAD / 2; ++c)
    P[row * SPAD + c0 + c] = (_Float16)(sc[row * SPAD + c0 + c] * invs);
  __syncthreads();

  // ---- O = P·V  (Vᵀ is [dim][seq] row-major -> X·Wᵀ form again) ----
  v8f o[8] = {};
  for (int ns = 0; ns < SPAD; ns += 32) {
    v16h ap = load_frag_a(P + ns, SPAD, lane);
#pragma unroll
    for (int dt = 0; dt < 8; ++dt) {
      v16h bv = load_frag_b(Vt + (long)(h * DHEAD + dt * 16) * SPAD + ns,
                            SPAD, lane);
      o[dt] = wmma_f16(ap, bv, o[dt]);
    }
  }
#pragma unroll
  for (int dt = 0; dt < 8; ++dt)
#pragma unroll
    for (int r = 0; r < 8; ++r) {
      const int m = q0 + half * 8 + r;
      O16[(long)m * DIM + h * DHEAD + dt * 16 + nc] = (_Float16)o[dt][r];
    }
}

// ---------------------------------------------------------------------------
// Launch
// ---------------------------------------------------------------------------
extern "C" void kernel_launch(void* const* d_in, const int* in_sizes, int n_in,
                              void* d_out, int out_size, void* d_ws,
                              size_t ws_size, hipStream_t stream) {
  (void)in_sizes; (void)n_in; (void)out_size; (void)ws_size;
  const float* hs   = (const float*)d_in[0];
  const float* fcos = (const float*)d_in[1];
  const float* fsin = (const float*)d_in[2];
  const float* wq   = (const float*)d_in[3];
  const float* bq   = (const float*)d_in[4];
  const float* wk   = (const float*)d_in[5];
  const float* bk   = (const float*)d_in[6];
  const float* wv   = (const float*)d_in[7];
  const float* bv   = (const float*)d_in[8];
  const float* nqw  = (const float*)d_in[9];
  const float* nkw  = (const float*)d_in[10];
  const float* wo   = (const float*)d_in[11];
  const float* bo   = (const float*)d_in[12];
  float* out = (float*)d_out;

  const size_t NH = (size_t)SPAD * DIM;  // halves per 2048x2048 f16 matrix
  _Float16* base = (_Float16*)d_ws;
  _Float16* X16  = base + 0 * NH;
  _Float16* Wq16 = base + 1 * NH;
  _Float16* Wk16 = base + 2 * NH;
  _Float16* Wv16 = base + 3 * NH;
  _Float16* Wo16 = base + 4 * NH;
  _Float16* Q16  = base + 5 * NH;
  _Float16* K16  = base + 6 * NH;
  _Float16* Vt16 = base + 7 * NH;
  _Float16* O16  = base + 8 * NH;

  // 0) pack f32 -> f16 (X padded, weights)
  pack_f16_kernel<<<dim3((SPAD * DIM) / (256 * 4), 5), 256, 0, stream>>>(
      hs, wq, wk, wv, wo, X16, Wq16, Wk16, Wv16, Wo16);

  // 1) QKV projections (V written transposed per-dim for the PV GEMM)
  const dim3 gg(SPAD / 128, DIM / 128);
  gemm_xwT_kernel<0><<<gg, 128, 0, stream>>>(X16, Wq16, bq, Q16, nullptr, SEQ);
  gemm_xwT_kernel<0><<<gg, 128, 0, stream>>>(X16, Wk16, bk, K16, nullptr, SEQ);
  gemm_xwT_kernel<1><<<gg, 128, 0, stream>>>(X16, Wv16, bv, Vt16, nullptr, SEQ);

  // 2) RMSNorm + RoPE on Q and K (in place)
  rmsnorm_rope_kernel<<<dim3(SEQ, 2), 256, 0, stream>>>(Q16, K16, nqw, nkw,
                                                        fcos, fsin);

  // 3) attention (dynamic LDS: 16x2048 f32 scores + 16x2048 f16 probs)
  const size_t lds_bytes = (size_t)16 * SPAD * 4 + (size_t)16 * SPAD * 2;
  attention_kernel<<<dim3(SPAD / 16, HEADS), 32, lds_bytes, stream>>>(
      Q16, K16, Vt16, O16);

  // 4) output projection -> f32 d_out
  gemm_xwT_kernel<2><<<gg, 128, 0, stream>>>(O16, Wo16, bo, nullptr, out, SEQ);
}
